// HeteroGNN_24060406792806
// MI455X (gfx1250) — matrix operand order repaired and checked by reading, not compiled, not run on hardware
//
#include <hip/hip_runtime.h>
#include <hip/hip_bf16.h>

// ---------------------------------------------------------------------------
// HeteroGNN on MI455X (gfx1250, wave32). Bandwidth-bound on edge scatter
// (~3.3 GB irregular traffic @ 23.3 TB/s); dense projections use
// v_wmma_f32_16x16x32_bf16 with algebraically pre-combined weights:
//   concat(xd@Wd+bd, agg@Ws+bs)@Wu + bu == xd@(Wd@Wu_top) + agg@(Ws@Wu_bot) + c
// Weights are pre-packed in B-fragment order (contiguous k = 16*half + s) so each lane
// loads its fragment with two global_load_b128.
// ---------------------------------------------------------------------------

#define NG_   50000
#define ND_   20000
#define Hdim  128
#define EGG_  800000
#define EGDA_ 400000
#define ELAB_ 200000

typedef __attribute__((ext_vector_type(16))) __bf16 v16bf;
typedef __attribute__((ext_vector_type(8)))  float  v8f;

union BFrag { v16bf bf; uint4 u[2]; };

// ----------------------------- utility kernels -----------------------------

__global__ void k_zero(float* __restrict__ p, long n) {
  for (long i = blockIdx.x * (long)blockDim.x + threadIdx.x; i < n;
       i += (long)gridDim.x * blockDim.x)
    p[i] = 0.0f;
}

__global__ void k_scatter_feat(const float* __restrict__ xsrc,
                               const int* __restrict__ src,
                               const int* __restrict__ dst,
                               float* __restrict__ agg, long E) {
  long total = E << 7;  // E * 128
  for (long i = blockIdx.x * (long)blockDim.x + threadIdx.x; i < total;
       i += (long)gridDim.x * blockDim.x) {
    long e = i >> 7;
    int  f = (int)(i & 127);
    atomicAdd(agg + (((size_t)dst[e]) << 7) + f, xsrc[(((size_t)src[e]) << 7) + f]);
  }
}

__global__ void k_deg(const int* __restrict__ dst, float* __restrict__ deg, long E) {
  for (long e = blockIdx.x * (long)blockDim.x + threadIdx.x; e < E;
       e += (long)gridDim.x * blockDim.x)
    atomicAdd(deg + dst[e], 1.0f);
}

__global__ void k_divdeg(float* __restrict__ agg, const float* __restrict__ deg, long N) {
  long total = N << 5;  // N * 32 float4s
  float4* a4 = (float4*)agg;
  for (long i = blockIdx.x * (long)blockDim.x + threadIdx.x; i < total;
       i += (long)gridDim.x * blockDim.x) {
    const float inv = 1.0f / fmaxf(deg[i >> 5], 1.0f);
    float4 v = a4[i];
    v.x *= inv; v.y *= inv; v.z *= inv; v.w *= inv;
    a4[i] = v;
  }
}

__global__ void k_combine(const float* __restrict__ a, const float* __restrict__ b,
                          float* __restrict__ o, long n4) {
  const float4* a4 = (const float4*)a;
  const float4* b4 = (const float4*)b;
  float4* o4 = (float4*)o;
  for (long i = blockIdx.x * (long)blockDim.x + threadIdx.x; i < n4;
       i += (long)gridDim.x * blockDim.x) {
    float4 x = a4[i], y = b4[i];
    o4[i] = make_float4(0.5f * (x.x + y.x), 0.5f * (x.y + y.y),
                        0.5f * (x.z + y.z), 0.5f * (x.w + y.w));
  }
}

// ------------------------- weight pre-combination --------------------------
// Packed layout per matrix j (8192 dwords): dword[((t*4+kc)*32 + lane)*8 + v]
// holds the bf16 pair (k, k+1) for tile t, k-chunk kc, given
//   lane = 16*(kk>>4) + (h&15), v = (kk&15)>>1, kk = k&31, kc = k>>5, t = h>>4.
// Each thread computes an even k and k+1 so the full dword has one writer.
__global__ void k_prep_w(const float* __restrict__ Wd1, const float* __restrict__ Ws1,
                         const float* __restrict__ Wu1, const float* __restrict__ Wd2,
                         const float* __restrict__ Ws2, const float* __restrict__ Wu2,
                         __bf16* __restrict__ Wc) {
  long total = 12L * 128 * 64;  // j * h * (k/2)
  for (long i = blockIdx.x * (long)blockDim.x + threadIdx.x; i < total;
       i += (long)gridDim.x * blockDim.x) {
    const int j  = (int)(i >> 13);
    const int r  = (int)(i & 8191);
    const int h  = r >> 6;
    const int k  = (r & 63) << 1;
    const int layer = j / 6, rem = j % 6, et = rem >> 1, part = rem & 1;
    const float* Wbase = (layer == 0 ? (part == 0 ? Wd1 : Ws1)
                                     : (part == 0 ? Wd2 : Ws2)) + (size_t)et * 128 * 128;
    const float* Wu = (layer == 0 ? Wu1 : Wu2) + (size_t)et * 256 * 128 + (size_t)part * 128 * 128;
    float s0 = 0.0f, s1 = 0.0f;
    for (int m = 0; m < 128; ++m) {
      const float u = Wu[m * 128 + h];
      s0 += Wbase[k * 128 + m] * u;
      s1 += Wbase[(k + 1) * 128 + m] * u;
    }
    const int t = h >> 4, nlo = h & 15;
    const int kc = k >> 5, kk = k & 31;
    const int lane = ((kk >> 4) << 4) + nlo;
    const int v = (kk & 15) >> 1;
    const long d = ((long)((t * 4 + kc) * 32 + lane)) * 8 + v;
    __bf16* out = Wc + (size_t)j * 16384 + d * 2;
    out[0] = (__bf16)s0;
    out[1] = (__bf16)s1;
  }
}

// bc[layer*3+et][h] = bd@Wu_top + bs@Wu_bot + bu
__global__ void k_prep_b(const float* __restrict__ bd1, const float* __restrict__ bs1,
                         const float* __restrict__ bu1, const float* __restrict__ Wu1,
                         const float* __restrict__ bd2, const float* __restrict__ bs2,
                         const float* __restrict__ bu2, const float* __restrict__ Wu2,
                         float* __restrict__ bc) {
  long total = 6L * 128;
  for (long i = blockIdx.x * (long)blockDim.x + threadIdx.x; i < total;
       i += (long)gridDim.x * blockDim.x) {
    int j = (int)(i >> 7), h = (int)(i & 127);
    int layer = j / 3, et = j % 3;
    const float* Wu = (layer == 0 ? Wu1 : Wu2) + (size_t)et * 256 * 128;
    const float* bd = (layer == 0 ? bd1 : bd2) + (size_t)et * 128;
    const float* bs = (layer == 0 ? bs1 : bs2) + (size_t)et * 128;
    const float* bu = (layer == 0 ? bu1 : bu2) + (size_t)et * 128;
    float s = bu[h];
    for (int m = 0; m < 128; ++m)
      s += bd[m] * Wu[m * 128 + h] + bs[m] * Wu[(128 + m) * 128 + h];
    bc[i] = s;
  }
}

// ------------------------------ WMMA GEMM ----------------------------------
// out[N x 128] = X @ WA + G @ WB + bias   (X,G fp32; WA,WB bf16 fragment-packed)
// 128 threads = 4 waves; wave owns 16 rows x 128 cols = 8 accumulator tiles.
__global__ __launch_bounds__(128)
void gemm2_wmma(const float* __restrict__ X, const float* __restrict__ G,
                const unsigned int* __restrict__ WA, const unsigned int* __restrict__ WB,
                const float* __restrict__ bias, float* __restrict__ out, int N) {
  const int lane    = threadIdx.x & 31;
  const int wave    = threadIdx.x >> 5;
  const int rowBase = blockIdx.x * 64 + wave * 16;
  const int mlo     = lane & 15;
  const int half    = lane >> 4;
  // Clamp: OOB rows read row N-1 (valid memory); their outputs are never stored.
  int arow = rowBase + mlo;
  arow = (arow < N) ? arow : (N - 1);

  v8f acc[8] = {};

  const float* Xrow = X + (size_t)arow * Hdim;
  const float* Grow = G + (size_t)arow * Hdim;

  #pragma unroll
  for (int srcSel = 0; srcSel < 2; ++srcSel) {
    const float* Arow = (srcSel == 0) ? Xrow : Grow;
    const unsigned int* W = (srcSel == 0) ? WA : WB;
    #pragma unroll
    for (int kc = 0; kc < 4; ++kc) {          // K = 128 in chunks of 32
      const int kbase = kc * 32;
      // A fragment (contiguous K order): 16 consecutive K floats per lane half.
      const float4* ap0 = (const float4*)(Arow + kbase + (half << 4));
      const float4* ap1 = (const float4*)(Arow + kbase + (half << 4) + 8);
      const float4 a0 = ap0[0], a1 = ap0[1];
      const float4 a2 = ap1[0], a3 = ap1[1];
      v16bf afrag;
      afrag[0]  = (__bf16)a0.x;  afrag[1]  = (__bf16)a0.y;
      afrag[2]  = (__bf16)a0.z;  afrag[3]  = (__bf16)a0.w;
      afrag[4]  = (__bf16)a1.x;  afrag[5]  = (__bf16)a1.y;
      afrag[6]  = (__bf16)a1.z;  afrag[7]  = (__bf16)a1.w;
      afrag[8]  = (__bf16)a2.x;  afrag[9]  = (__bf16)a2.y;
      afrag[10] = (__bf16)a2.z;  afrag[11] = (__bf16)a2.w;
      afrag[12] = (__bf16)a3.x;  afrag[13] = (__bf16)a3.y;
      afrag[14] = (__bf16)a3.z;  afrag[15] = (__bf16)a3.w;
      #pragma unroll
      for (int t = 0; t < 8; ++t) {           // 8 column tiles of 16
        // Fragment-packed B: 8 consecutive dwords per lane -> 2x b128 loads.
        const uint4* wp = (const uint4*)(W + ((size_t)((t * 4 + kc) * 32 + lane) << 3));
        BFrag bf;
        bf.u[0] = wp[0];
        bf.u[1] = wp[1];
        acc[t] = __builtin_amdgcn_wmma_f32_16x16x32_bf16(
            false, afrag, false, bf.bf, (short)0, acc[t], false, false);
      }
    }
  }

  // Epilogue: C/D layout — lane col n = lane&15, VGPR v row m = 8*half + v
  if (rowBase + 16 <= N) {                    // wave-uniform full-tile fast path
    #pragma unroll
    for (int t = 0; t < 8; ++t) {
      const int n  = t * 16 + mlo;
      const float bv = bias[n];
      #pragma unroll
      for (int v = 0; v < 8; ++v) {
        const int orow = rowBase + (half << 3) + v;
        out[(size_t)orow * Hdim + n] = acc[t][v] + bv;
      }
    }
  } else {
    #pragma unroll
    for (int t = 0; t < 8; ++t) {
      const int n  = t * 16 + mlo;
      const float bv = bias[n];
      #pragma unroll
      for (int v = 0; v < 8; ++v) {
        const int orow = rowBase + (half << 3) + v;
        if (orow < N) out[(size_t)orow * Hdim + n] = acc[t][v] + bv;
      }
    }
  }
}

// ------------------------------- batchnorm ---------------------------------

__global__ __launch_bounds__(128)
void k_colstats(const float* __restrict__ x, float* __restrict__ stats, int N) {
  const int col = threadIdx.x;   // blockDim.x == 128
  float s = 0.0f, s2 = 0.0f;
  for (int r = blockIdx.x; r < N; r += gridDim.x) {
    const float v = x[(size_t)r * Hdim + col];
    s += v; s2 += v * v;
  }
  atomicAdd(&stats[col], s);
  atomicAdd(&stats[128 + col], s2);
}

__global__ void k_bn_lrelu(float* __restrict__ x, const float* __restrict__ stats,
                           const float* __restrict__ gamma, const float* __restrict__ beta,
                           int N) {
  const float invN = 1.0f / (float)N;
  long total = (long)N << 5;    // N * 32 float4s
  float4* x4 = (float4*)x;
  for (long i = blockIdx.x * (long)blockDim.x + threadIdx.x; i < total;
       i += (long)gridDim.x * blockDim.x) {
    const int c0 = (int)((i << 2) & 127);
    float4 v = x4[i];
    float* vp = &v.x;
    #pragma unroll
    for (int c = 0; c < 4; ++c) {
      const int col = c0 + c;
      const float m = stats[col] * invN;
      const float var = stats[128 + col] * invN - m * m;
      float y = (vp[c] - m) * rsqrtf(var + 1e-5f) * gamma[col] + beta[col];
      vp[c] = (y >= 0.0f) ? y : 0.01f * y;
    }
    x4[i] = v;
  }
}

// -------------------------------- decoder ----------------------------------

__global__ __launch_bounds__(256)
void k_decode(const float* __restrict__ g, const float* __restrict__ d,
              const int* __restrict__ ls, const int* __restrict__ ld,
              float* __restrict__ out, int E) {
  const int lane = threadIdx.x & 31;
  const int wid  = (int)((blockIdx.x * (long)blockDim.x + threadIdx.x) >> 5);
  if (wid >= E) return;
  const float4* gp = (const float4*)(g + ((size_t)ls[wid] << 7));
  const float4* dp = (const float4*)(d + ((size_t)ld[wid] << 7));
  const float4 gv = gp[lane];
  const float4 dv = dp[lane];
  float s = gv.x * dv.x + gv.y * dv.y + gv.z * dv.z + gv.w * dv.w;
  #pragma unroll
  for (int off = 16; off > 0; off >>= 1) s += __shfl_xor(s, off, 32);
  if (lane == 0) out[wid] = s;
}

// ------------------------------ host driver --------------------------------

static void run_conv(const float* x_src_nodes, const float* x_dst_nodes,
                     const int* src_idx, const int* dst_idx, long E, int Ndst,
                     const unsigned int* WA, const unsigned int* WB, const float* bias,
                     float* agg, float* deg, float* out, hipStream_t stream) {
  k_zero<<<1024, 256, 0, stream>>>(agg, (long)Ndst * Hdim);
  k_zero<<<128, 256, 0, stream>>>(deg, (long)Ndst);
  k_scatter_feat<<<4096, 256, 0, stream>>>(x_src_nodes, src_idx, dst_idx, agg, E);
  k_deg<<<1024, 256, 0, stream>>>(dst_idx, deg, E);
  k_divdeg<<<2048, 256, 0, stream>>>(agg, deg, (long)Ndst);
  gemm2_wmma<<<(Ndst + 63) / 64, 128, 0, stream>>>(x_dst_nodes, agg, WA, WB, bias, out, Ndst);
}

extern "C" void kernel_launch(void* const* d_in, const int* in_sizes, int n_in,
                              void* d_out, int out_size, void* d_ws, size_t ws_size,
                              hipStream_t stream) {
  (void)in_sizes; (void)n_in; (void)out_size; (void)ws_size;

  const float* x_gene = (const float*)d_in[0];
  const float* x_dis  = (const float*)d_in[1];
  const float* W_dst1 = (const float*)d_in[2];
  const float* W_src1 = (const float*)d_in[3];
  const float* W_upd1 = (const float*)d_in[4];
  const float* b_dst1 = (const float*)d_in[5];
  const float* b_src1 = (const float*)d_in[6];
  const float* b_upd1 = (const float*)d_in[7];
  const float* W_dst2 = (const float*)d_in[8];
  const float* W_src2 = (const float*)d_in[9];
  const float* W_upd2 = (const float*)d_in[10];
  const float* b_dst2 = (const float*)d_in[11];
  const float* b_src2 = (const float*)d_in[12];
  const float* b_upd2 = (const float*)d_in[13];
  const float* bn_gamma = (const float*)d_in[14];
  const float* bn_beta  = (const float*)d_in[15];
  const int* gg_src  = (const int*)d_in[16];
  const int* gg_dst  = (const int*)d_in[17];
  const int* gda_src = (const int*)d_in[18];
  const int* gda_dst = (const int*)d_in[19];
  const int* label_src = (const int*)d_in[20];
  const int* label_dst = (const int*)d_in[21];

  const size_t fNG = (size_t)NG_ * Hdim;
  const size_t fND = (size_t)ND_ * Hdim;

  float* agg   = (float*)d_ws;
  float* tA    = agg + fNG;
  float* tB    = tA + fNG;
  float* g1    = tB + fNG;          // layer-1 gene out; reused as g2
  float* d1    = g1 + fNG;
  float* d2    = d1 + fND;
  float* deg   = d2 + fND;          // NG_ floats
  float* bc    = deg + NG_;         // 6 * 128
  float* stats = bc + 6 * 128;      // 256
  __bf16* Wc   = (__bf16*)(stats + 256);  // 12 * 16384 bf16, fragment-packed

  // 0) pre-combine weights / biases (bf16 weights packed in B-fragment order)
  k_prep_w<<<384, 256, 0, stream>>>(W_dst1, W_src1, W_upd1, W_dst2, W_src2, W_upd2, Wc);
  k_prep_b<<<3, 256, 0, stream>>>(b_dst1, b_src1, b_upd1, W_upd1,
                                  b_dst2, b_src2, b_upd2, W_upd2, bc);

  #define WCJ(layer, et, part) \
    ((const unsigned int*)(Wc + ((size_t)((layer) * 6 + (et) * 2 + (part))) * 16384))
  #define BCJ(layer, et) (bc + ((layer) * 3 + (et)) * 128)

  // 1) layer 1
  run_conv(x_gene, x_gene, gg_src,  gg_dst,  EGG_,  NG_, WCJ(0,0,0), WCJ(0,0,1), BCJ(0,0), agg, deg, tA, stream);
  run_conv(x_gene, x_dis,  gda_src, gda_dst, EGDA_, ND_, WCJ(0,1,0), WCJ(0,1,1), BCJ(0,1), agg, deg, d1, stream);
  run_conv(x_dis,  x_gene, gda_dst, gda_src, EGDA_, NG_, WCJ(0,2,0), WCJ(0,2,1), BCJ(0,2), agg, deg, tB, stream);
  k_combine<<<2048, 256, 0, stream>>>(tA, tB, g1, (long)(fNG >> 2));

  // 2) batchnorm + leaky relu (training-mode batch stats)
  k_zero<<<1, 256, 0, stream>>>(stats, 256);
  k_colstats<<<256, 128, 0, stream>>>(g1, stats, NG_);
  k_bn_lrelu<<<2048, 256, 0, stream>>>(g1, stats, bn_gamma, bn_beta, NG_);
  k_zero<<<1, 256, 0, stream>>>(stats, 256);
  k_colstats<<<256, 128, 0, stream>>>(d1, stats, ND_);
  k_bn_lrelu<<<2048, 256, 0, stream>>>(d1, stats, bn_gamma + 128, bn_beta + 128, ND_);

  // 3) layer 2 (g2 aliases g1 after convs complete)
  run_conv(g1, g1, gg_src,  gg_dst,  EGG_,  NG_, WCJ(1,0,0), WCJ(1,0,1), BCJ(1,0), agg, deg, tA, stream);
  run_conv(g1, d1, gda_src, gda_dst, EGDA_, ND_, WCJ(1,1,0), WCJ(1,1,1), BCJ(1,1), agg, deg, d2, stream);
  run_conv(d1, g1, gda_dst, gda_src, EGDA_, NG_, WCJ(1,2,0), WCJ(1,2,1), BCJ(1,2), agg, deg, tB, stream);
  k_combine<<<2048, 256, 0, stream>>>(tA, tB, g1, (long)(fNG >> 2));

  // 4) dot-product decoder on label edges (one wave per edge)
  k_decode<<<(ELAB_ * 32 + 255) / 256, 256, 0, stream>>>(g1, d2, label_src, label_dst,
                                                         (float*)d_out, ELAB_);
  #undef WCJ
  #undef BCJ
}
